// _ScaledDotProductAttention_59914793779824
// MI455X (gfx1250) — matrix operand order repaired and checked by reading, not compiled, not run on hardware
//
#include <hip/hip_runtime.h>
#include <hip/hip_bf16.h>
#include <stdint.h>

typedef __bf16 v16bf __attribute__((ext_vector_type(16)));
typedef float  v8f   __attribute__((ext_vector_type(8)));

#define B_   16
#define S_   2048
#define D_   128
#define QT   16     // query rows per workgroup
#define NWAVE 4
#define NTHREADS 128

// LDS layout (floats): probbuf[16*2048] | outbuf[16*128] | wavemax[4*16] | rowmax[16] | rowsum[16] | rowinv[16]
#define SMEM_FLOATS (QT*S_ + QT*D_ + NWAVE*QT + QT + QT + QT)

__global__ void attn_fwd_gfx1250(const float* __restrict__ Q,
                                 const float* __restrict__ K,
                                 const float* __restrict__ V,
                                 float* __restrict__ Out,
                                 float* __restrict__ Attn)
{
    extern __shared__ float smem[];
    float* probbuf = smem;                  // [16][2048] raw scores -> exp -> normalized probs
    float* outbuf  = smem + QT * S_;        // [16][128] output accumulation
    float* wavemax = outbuf + QT * D_;      // [4][16]
    float* rowmax  = wavemax + NWAVE * QT;  // [16]
    float* rowsum  = rowmax + QT;           // [16]
    float* rowinv  = rowsum + QT;           // [16]

    const int tid  = threadIdx.x;
    const int w    = tid >> 5;        // wave id 0..3
    const int lane = tid & 31;
    const int nl   = lane & 15;
    const int hi   = (lane >> 4) & 1; // upper half of wave

    const int q0   = blockIdx.x * QT;
    const int bidx = blockIdx.y;
    const int qend = q0 + QT - 1;

    const float rscale = 0.0883883476483184f; // 1/sqrt(128)

    // zero the output accumulation buffer
    #pragma unroll
    for (int i = 0; i < (QT * D_) / NTHREADS; ++i)
        outbuf[i * NTHREADS + tid] = 0.0f;

    const float* qb = Q + ((size_t)bidx * S_ + q0) * D_;
    const float* kb = K + (size_t)bidx * S_ * D_;
    const float* vb = V + (size_t)bidx * S_ * D_;

    // ---- Q tile -> WMMA A-layout bf16 regs; 4 k-steps of 32 along D ----
    // lane<16 : M=nl, elems 0..7 = d[32s+0..7],  8..15 = d[32s+16..23]
    // lane>=16: M=nl, elems 0..7 = d[32s+8..15], 8..15 = d[32s+24..31]
    v16bf qreg[4];
    {
        const float* qrow = qb + nl * D_;
        #pragma unroll
        for (int s = 0; s < 4; ++s) {
            const float* p = qrow + 32 * s + (hi ? 8 : 0);
            v16bf a;
            #pragma unroll
            for (int i = 0; i < 8; ++i) a[i]     = (__bf16)p[i];
            #pragma unroll
            for (int i = 0; i < 8; ++i) a[8 + i] = (__bf16)p[16 + i];
            qreg[s] = a;
        }
    }

    // ---- pass 1: S = (Q K^T) * rscale -> LDS; track per-row max (masked) ----
    float runmax[8];
    #pragma unroll
    for (int r = 0; r < 8; ++r) runmax[r] = -3.0e38f;

    for (int c = w; c * 32 <= qend; c += NWAVE) {
        #pragma unroll
        for (int half = 0; half < 2; ++half) {
            const int k0 = c * 32 + half * 16;
            if (k0 > qend) continue;
            v8f acc = {};
            #pragma unroll
            for (int s = 0; s < 4; ++s) {
                // B tile (K^T): lane<16: N=nl, d=32s+0..7,16..23 ; lane>=16: N=nl, d=32s+8..15,24..31
                const float* p = kb + (size_t)(k0 + nl) * D_ + 32 * s + (hi ? 8 : 0);
                v16bf bk;
                #pragma unroll
                for (int i = 0; i < 16; ++i) bk[i] = (__bf16)p[(i < 8) ? i : i + 8];
                acc = __builtin_amdgcn_wmma_f32_16x16x32_bf16(
                        false, qreg[s], false, bk, (short)0, acc, false, false);
            }
            // C layout: vgpr r, lane -> row m = r + 8*hi, col n = nl
            #pragma unroll
            for (int r = 0; r < 8; ++r) {
                const int m  = r + (hi ? 8 : 0);
                const int kk = k0 + nl;
                const float sc = acc[r] * rscale;
                probbuf[m * S_ + kk] = sc;
                float mv = (kk <= q0 + m) ? sc : -3.0e38f;
                #pragma unroll
                for (int off = 1; off < 16; off <<= 1)
                    mv = fmaxf(mv, __shfl_xor(mv, off, 32));
                runmax[r] = fmaxf(runmax[r], mv);
            }
        }
    }

    if (nl == 0) {
        #pragma unroll
        for (int r = 0; r < 8; ++r)
            wavemax[w * QT + r + (hi ? 8 : 0)] = runmax[r];
    }
    __syncthreads();

    if (tid < QT) {
        float mx = wavemax[tid];
        #pragma unroll
        for (int wv = 1; wv < NWAVE; ++wv) mx = fmaxf(mx, wavemax[wv * QT + tid]);
        rowmax[tid] = mx;
        rowsum[tid] = 0.0f;
    }
    __syncthreads();

    // ---- pass 2: p = exp(s - max) with causal mask, row sums, normalize ----
    {
        const int m    = tid >> 3;          // 8 threads per row
        const int cb   = (tid & 7) * 256;   // 256 contiguous cols each
        const float rm = rowmax[m];
        const int qrow = q0 + m;
        float sum = 0.0f;
        for (int kk = cb; kk < cb + 256; kk += 4) {
            float4 sv = *(float4*)&probbuf[m * S_ + kk];
            float p0 = (kk + 0 <= qrow) ? __expf(sv.x - rm) : 0.0f;
            float p1 = (kk + 1 <= qrow) ? __expf(sv.y - rm) : 0.0f;
            float p2 = (kk + 2 <= qrow) ? __expf(sv.z - rm) : 0.0f;
            float p3 = (kk + 3 <= qrow) ? __expf(sv.w - rm) : 0.0f;
            sum += (p0 + p1) + (p2 + p3);
            *(float4*)&probbuf[m * S_ + kk] = make_float4(p0, p1, p2, p3);
        }
        atomicAdd(&rowsum[m], sum);         // ds_add_f32
        __syncthreads();
        if (tid < QT) rowinv[tid] = 1.0f / rowsum[tid];
        __syncthreads();
        const float inv = rowinv[m];
        for (int kk = cb; kk < cb + 256; kk += 4) {
            float4 pv = *(float4*)&probbuf[m * S_ + kk];
            pv.x *= inv; pv.y *= inv; pv.z *= inv; pv.w *= inv;
            *(float4*)&probbuf[m * S_ + kk] = pv;
        }
    }
    __syncthreads();

    // ---- async LDS -> global copy of the attn block (contiguous 128 KB) ----
    // attn[b][q0..q0+15][0..2047] is one contiguous region; probbuf matches it.
    {
        const uint64_t sbase = (uint64_t)(uintptr_t)Attn;
        const uint32_t goff0 = (uint32_t)((((size_t)bidx * S_ + q0) * (size_t)S_) * sizeof(float));
        #pragma unroll 4
        for (int it = 0; it < (QT * S_) / (NTHREADS * 4); ++it) {  // 64 iters of b128/lane
            const int e = it * NTHREADS + tid;                     // float4 index
            const uint32_t voff = goff0 + (uint32_t)e * 16u;
            const uint32_t ldsa = (uint32_t)(uintptr_t)&probbuf[e * 4];
            asm volatile("global_store_async_from_lds_b128 %0, %1, %2"
                         :: "v"(voff), "v"(ldsa), "s"(sbase) : "memory");
        }
    }

    // ---- pass 3 (overlaps async stores): Out = P @ V via bf16 WMMA ----
    v8f accd[8] = {};
    for (int c = w; c * 32 <= qend; c += NWAVE) {
        const int kc = c * 32;
        // A tile = normalized probs, 16 rows x 32 keys, from LDS
        const float* pr = &probbuf[nl * S_ + kc + (hi ? 8 : 0)];
        v16bf aP;
        #pragma unroll
        for (int i = 0; i < 8; ++i) aP[i]     = (__bf16)pr[i];
        #pragma unroll
        for (int i = 0; i < 8; ++i) aP[8 + i] = (__bf16)pr[16 + i];
        // B tiles = V chunk (32 keys x 16 d-cols per n-tile)
        const float* vrow = vb + (size_t)(kc + (hi ? 8 : 0)) * D_ + nl;
        #pragma unroll
        for (int n = 0; n < 8; ++n) {
            const float* vp = vrow + n * 16;
            v16bf bV;
            #pragma unroll
            for (int i = 0; i < 16; ++i) bV[i] = (__bf16)vp[(size_t)((i < 8) ? i : i + 8) * D_];
            accd[n] = __builtin_amdgcn_wmma_f32_16x16x32_bf16(
                        false, aP, false, bV, (short)0, accd[n], false, false);
        }
    }
    // reduce the 4 waves' partial outputs through LDS
    #pragma unroll
    for (int n = 0; n < 8; ++n) {
        #pragma unroll
        for (int r = 0; r < 8; ++r)
            atomicAdd(&outbuf[(r + (hi ? 8 : 0)) * D_ + n * 16 + nl], accd[n][r]);
    }
    __syncthreads();

    // ---- coalesced out write (16x128 contiguous block) ----
    {
        float* ob = Out + ((size_t)bidx * S_ + q0) * D_;
        #pragma unroll
        for (int i = 0; i < (QT * D_) / NTHREADS; ++i) {
            const int idx = i * NTHREADS + tid;
            __builtin_nontemporal_store(outbuf[idx], &ob[idx]);
        }
    }

    // drain async LDS->global stores before wave exit
    asm volatile("s_wait_asynccnt 0x0" ::: "memory");
}

extern "C" void kernel_launch(void* const* d_in, const int* in_sizes, int n_in,
                              void* d_out, int out_size, void* d_ws, size_t ws_size,
                              hipStream_t stream) {
    (void)in_sizes; (void)n_in; (void)d_ws; (void)ws_size; (void)out_size;
    const float* q = (const float*)d_in[0];
    const float* k = (const float*)d_in[1];
    const float* v = (const float*)d_in[2];
    // d_in[3] is the causal mask; causality is computed analytically in-kernel.
    float* out  = (float*)d_out;
    float* attn = out + (size_t)B_ * S_ * D_;   // outputs concatenated: out, attn

    dim3 grid(S_ / QT, B_);
    const size_t smem = (size_t)SMEM_FLOATS * sizeof(float);  // ~136.5 KB dynamic LDS (<= 320 KB/WGP)
    attn_fwd_gfx1250<<<grid, NTHREADS, smem, stream>>>(q, k, v, out, attn);
}
